// MambaMLPLayer_88356067213971
// MI455X (gfx1250) — compile-verified, matrix operand order checked by reading, not run on hardware
//
#include <hip/hip_runtime.h>

typedef __bf16 bf16;
typedef __attribute__((ext_vector_type(16))) __bf16 v16bf;
typedef __attribute__((ext_vector_type(8)))  __bf16 v8bf;
typedef __attribute__((ext_vector_type(8)))  float  v8f;
typedef unsigned int u32;
typedef unsigned long long u64;
typedef __attribute__((ext_vector_type(4))) u32 u32x4;
typedef __attribute__((ext_vector_type(8))) u32 u32x8;

union AFrag { v16bf v; v8bf h[2]; };

constexpr int BQ = 2, LQ = 1024, DIMQ = 512, DINQ = 2048, DTR = 32;
constexpr int MROWS = BQ * LQ;           // 2048 token rows

// ---------------------------------------------------------------------------
// fp32 -> bf16 cast (weights, done once per launch since no caching allowed)
// ---------------------------------------------------------------------------
__global__ __launch_bounds__(256) void k_cast(const float* __restrict__ s,
                                              bf16* __restrict__ d, int n) {
  int i = blockIdx.x * 256 + threadIdx.x;
  if (i < n) d[i] = (bf16)s[i];
}

// ---------------------------------------------------------------------------
// LayerNorm over DIM=512, one wave32 per row (16 cols/lane), bf16 output
// ---------------------------------------------------------------------------
__global__ __launch_bounds__(256) void k_layernorm_cast(
    const float* __restrict__ in, const float* __restrict__ g,
    const float* __restrict__ b, bf16* __restrict__ out, float eps) {
  const int lane = threadIdx.x & 31;
  const int row = blockIdx.x * 8 + (threadIdx.x >> 5);
  const float* x = in + (u64)row * DIMQ;
  float vals[16];
  float s = 0.f;
#pragma unroll
  for (int j = 0; j < 16; ++j) { vals[j] = x[j * 32 + lane]; s += vals[j]; }
  for (int off = 16; off; off >>= 1) s += __shfl_xor(s, off, 32);
  const float mean = s * (1.0f / DIMQ);
  float vs = 0.f;
#pragma unroll
  for (int j = 0; j < 16; ++j) { float dv = vals[j] - mean; vs += dv * dv; }
  for (int off = 16; off; off >>= 1) vs += __shfl_xor(vs, off, 32);
  const float inv = rsqrtf(vs * (1.0f / DIMQ) + eps);
  bf16* o = out + (u64)row * DIMQ;
#pragma unroll
  for (int j = 0; j < 16; ++j) {
    int c = j * 32 + lane;
    o[c] = (bf16)((vals[j] - mean) * inv * g[c] + b[c]);
  }
}

// ---------------------------------------------------------------------------
// Generic bf16 WMMA GEMM: C[M,N] = A[M,K] * W[N,K]^T  (both K-contiguous)
// Wave computes a 16x32 C strip (two 16x16 tiles sharing the A fragment);
// block = 8 waves = 32(M) x 128(N). A panel (32 x Kchunk<=512) staged into
// LDS by the Tensor Data Mover. Inner loop register-double-buffers fragments
// so loads for step k+32 overlap the two WMMAs of step k.
// Epilogue: +bias -> act(0 none,1 softplus,2 gelu) -> +residual -> f32/bf16.
// ---------------------------------------------------------------------------
__global__ __launch_bounds__(256) void k_gemm_bf16(
    const bf16* __restrict__ A, const bf16* __restrict__ W, int N, int K,
    const float* __restrict__ bias, const float* __restrict__ resid, int act,
    float* __restrict__ outF, bf16* __restrict__ outB) {
  __shared__ __align__(16) bf16 sA[32 * 512];
  const int tid = threadIdx.x;
  const int lane = tid & 31;
  const int wid = tid >> 5;
  const int wm = wid >> 2;                 // 0..1
  const int wn = wid & 3;                  // 0..3
  const int m0 = blockIdx.y * 32 + wm * 16;
  const int n0 = (blockIdx.x * 8 + wn * 2) * 16;   // two adjacent N tiles
  const bool act0 = (n0 < N);
  const bool act1 = (n0 + 16 < N);
  const int r = lane & 15;                 // row within frag / col within tile
  const int kh = lane >> 4;                // K-half select per ISA lane layout

  v8f acc0 = {}, acc1 = {};

  for (int kb = 0; kb < K; kb += 512) {
    const int chunk = (K - kb < 512) ? (K - kb) : 512;
    __syncthreads();                        // previous chunk fully consumed
    if (tid == 0) {
      // Tensor DMA descriptor: load 32 x chunk bf16 tile (row stride K) to LDS
      u64 ga = (u64)A + ((u64)(blockIdx.y * 32) * (u64)K + (u64)kb) * 2ull;
      u32 lds_off = (u32)(u64)(&sA[0]);
      u32x4 g0;
      g0[0] = 1u;                                        // count=1, user mode
      g0[1] = lds_off;                                   // lds_addr
      g0[2] = (u32)ga;                                   // global_addr lo
      g0[3] = ((u32)(ga >> 32) & 0x01FFFFFFu) | (2u << 30); // addr hi | type=2
      u32x8 g1;
      g1[0] = 0x10000u;                                  // data_size=1 (2B)
      g1[1] = ((u32)K) << 16;                            // tensor_dim0 lo16
      g1[2] = (((u32)K) >> 16) | (((u32)MROWS) << 16);   // dim0 hi | dim1 lo
      g1[3] = (((u32)MROWS) >> 16) | (((u32)chunk) << 16); // dim1 hi|tile_dim0
      g1[4] = 32u;                                       // tile_dim1 = 32 rows
      g1[5] = (u32)K;                                    // tensor_dim0_stride
      g1[6] = 0u;
      g1[7] = 0u;
      asm volatile("tensor_load_to_lds %0, %1" ::"s"(g0), "s"(g1) : "memory");
      __builtin_amdgcn_s_wait_tensorcnt(0);
    }
    __syncthreads();

    if (act0) {
      const bf16* w0 = W + (u64)(n0 + r) * K + kb;
      const bf16* w1p = W + (u64)(n0 + 16 + r) * K + kb;
      __builtin_prefetch(w0 + chunk, 0, 0);              // global_prefetch_b8
      const bf16* arow = &sA[(wm * 16 + r) * chunk];

      AFrag a, b0, b1;
      a.h[0] = *(const v8bf*)(arow + kh * 8);
      a.h[1] = *(const v8bf*)(arow + 16 + kh * 8);
      b0.h[0] = *(const v8bf*)(w0 + kh * 8);
      b0.h[1] = *(const v8bf*)(w0 + 16 + kh * 8);
      if (act1) {
        b1.h[0] = *(const v8bf*)(w1p + kh * 8);
        b1.h[1] = *(const v8bf*)(w1p + 16 + kh * 8);
      }
      for (int kk = 0; kk < chunk; kk += 32) {
        // double-buffer: issue next fragment loads before this step's WMMAs
        const int kn = (kk + 32 < chunk) ? kk + 32 : kk;
        AFrag an, bn0, bn1;
        an.h[0] = *(const v8bf*)(arow + kn + kh * 8);
        an.h[1] = *(const v8bf*)(arow + kn + 16 + kh * 8);
        bn0.h[0] = *(const v8bf*)(w0 + kn + kh * 8);
        bn0.h[1] = *(const v8bf*)(w0 + kn + 16 + kh * 8);
        if (act1) {
          bn1.h[0] = *(const v8bf*)(w1p + kn + kh * 8);
          bn1.h[1] = *(const v8bf*)(w1p + kn + 16 + kh * 8);
        }
        acc0 = __builtin_amdgcn_wmma_f32_16x16x32_bf16(
            false, a.v, false, b0.v, (short)0, acc0, false, false);
        if (act1)
          acc1 = __builtin_amdgcn_wmma_f32_16x16x32_bf16(
              false, a.v, false, b1.v, (short)0, acc1, false, false);
        a = an; b0 = bn0; b1 = bn1;
      }
    }
  }

  if (act0) {
    const int col0 = n0 + r;
    const int col1 = n0 + 16 + r;
    const float bv0 = bias ? bias[col0] : 0.0f;
    const float bv1 = (bias && act1) ? bias[col1] : 0.0f;
#pragma unroll
    for (int v = 0; v < 8; ++v) {
      const int row = m0 + v + (kh << 3);   // lanes 16-31 hold M = v+8
      float c0 = acc0[v] + bv0;
      float c1 = acc1[v] + bv1;
      if (act == 1) {
        c0 = (c0 > 20.0f) ? c0 : log1pf(__expf(c0));
        c1 = (c1 > 20.0f) ? c1 : log1pf(__expf(c1));
      } else if (act == 2) {
        c0 = 0.5f * c0 * (1.0f + erff(c0 * 0.70710678f));
        c1 = 0.5f * c1 * (1.0f + erff(c1 * 0.70710678f));
      }
      const u64 i0 = (u64)row * (u64)N + (u64)col0;
      const u64 i1 = (u64)row * (u64)N + (u64)col1;
      if (resid) {
        c0 += resid[i0];
        if (act1) c1 += resid[i1];
      }
      if (outF) {
        outF[i0] = c0;
        if (act1) outF[i1] = c1;
      }
      if (outB) {
        outB[i0] = (bf16)c0;
        if (act1) outB[i1] = (bf16)c1;
      }
    }
  }
}

// ---------------------------------------------------------------------------
// Depthwise causal conv (D_CONV=4) + SiLU over u = xz[:,:, :2048]
// ---------------------------------------------------------------------------
__global__ __launch_bounds__(256) void k_conv_silu(
    const float* __restrict__ xz, const float* __restrict__ cw,
    const float* __restrict__ cb, float* __restrict__ uF,
    bf16* __restrict__ uB) {
  const int idx = blockIdx.x * 256 + threadIdx.x;   // b(1) l(10) d(11)
  const int d = idx & (DINQ - 1);
  const int l = (idx >> 11) & (LQ - 1);
  const int b = idx >> 21;
  float acc = cb[d];
#pragma unroll
  for (int k = 0; k < 4; ++k) {
    const int ls = l + k - 3;
    if (ls >= 0) acc += cw[d * 4 + k] * xz[(u64)(b * LQ + ls) * 4096 + d];
  }
  const float s = acc / (1.0f + __expf(-acc));
  uF[idx] = s;
  uB[idx] = (bf16)s;
}

// x_dbl[:, :32] slice -> bf16 for the dt GEMM
__global__ __launch_bounds__(256) void k_dtslice(const float* __restrict__ xdbl,
                                                 bf16* __restrict__ dt) {
  const int idx = blockIdx.x * 256 + threadIdx.x;   // 2048*32
  dt[idx] = (bf16)xdbl[(idx >> 5) * 160 + (idx & 31)];
}

// ---------------------------------------------------------------------------
// Selective scan: one wave32 per (b,d) channel; 64 states = 2 per lane.
// Fused epilogue: ys = (scan_y + u*D) * silu(z), stored bf16 for out_proj.
// ---------------------------------------------------------------------------
__global__ __launch_bounds__(256) void k_scan(
    const float* __restrict__ dtv, const float* __restrict__ xdbl,
    const float* __restrict__ uF, const float* __restrict__ xz,
    const float* __restrict__ A_log, const float* __restrict__ Dp,
    bf16* __restrict__ ys) {
  const int lane = threadIdx.x & 31;
  const int gid = blockIdx.x * 8 + (threadIdx.x >> 5);  // 0..4095
  const int b = gid >> 11;
  const int d = gid & (DINQ - 1);
  const float A0 = -__expf(A_log[d * 64 + lane]);
  const float A1 = -__expf(A_log[d * 64 + lane + 32]);
  const float Dd = Dp[d];
  float h0 = 0.f, h1 = 0.f;
  const u64 rowbase = (u64)b * LQ;
  for (int t = 0; t < LQ; ++t) {
    const u64 rl = rowbase + t;
    const float dt_t = dtv[rl * DINQ + d];
    const float u_t = uF[rl * DINQ + d];
    const float* xr = xdbl + rl * 160;
    const float B0 = xr[32 + lane], B1 = xr[64 + lane];
    const float C0 = xr[96 + lane], C1 = xr[128 + lane];
    const float du = dt_t * u_t;
    h0 = __expf(dt_t * A0) * h0 + du * B0;
    h1 = __expf(dt_t * A1) * h1 + du * B1;
    float y = h0 * C0 + h1 * C1;
    for (int off = 16; off; off >>= 1) y += __shfl_xor(y, off, 32);
    if (lane == 0) {
      const float zv = xz[rl * 4096 + DINQ + d];
      const float sz = zv / (1.0f + __expf(-zv));
      ys[rl * DINQ + d] = (bf16)((y + u_t * Dd) * sz);
    }
  }
}

// ---------------------------------------------------------------------------
extern "C" void kernel_launch(void* const* d_in, const int* in_sizes, int n_in,
                              void* d_out, int out_size, void* d_ws,
                              size_t ws_size, hipStream_t stream) {
  (void)in_sizes; (void)n_in; (void)out_size; (void)ws_size;
  const float* x        = (const float*)d_in[0];
  const float* ln1_g    = (const float*)d_in[1];
  const float* ln1_b    = (const float*)d_in[2];
  const float* in_proj  = (const float*)d_in[3];
  const float* conv_w   = (const float*)d_in[4];
  const float* conv_b   = (const float*)d_in[5];
  const float* x_proj   = (const float*)d_in[6];
  const float* dt_w     = (const float*)d_in[7];
  const float* dt_b     = (const float*)d_in[8];
  const float* A_log    = (const float*)d_in[9];
  const float* Dp       = (const float*)d_in[10];
  const float* out_proj = (const float*)d_in[11];
  const float* ln2_g    = (const float*)d_in[12];
  const float* ln2_b    = (const float*)d_in[13];
  const float* w1       = (const float*)d_in[14];
  const float* b1       = (const float*)d_in[15];
  const float* w2       = (const float*)d_in[16];
  const float* b2       = (const float*)d_in[17];

  char* p = (char*)d_ws;
  auto take = [&](size_t bytes) {
    char* r = p; p += (bytes + 255) & ~(size_t)255; return r;
  };
  float* xz   = (float*)take((size_t)MROWS * 4096 * 4);
  float* uF   = (float*)take((size_t)MROWS * DINQ * 4);
  float* xdbl = (float*)take((size_t)MROWS * 160 * 4);
  float* dtv  = (float*)take((size_t)MROWS * DINQ * 4);
  float* x2   = (float*)take((size_t)MROWS * DIMQ * 4);
  bf16*  yb   = (bf16*)take((size_t)MROWS * DIMQ * 2);
  bf16*  uB   = (bf16*)take((size_t)MROWS * DINQ * 2);
  bf16*  dtin = (bf16*)take((size_t)MROWS * DTR * 2);
  bf16*  ysB  = (bf16*)take((size_t)MROWS * DINQ * 2);
  bf16*  hB   = (bf16*)take((size_t)MROWS * DIMQ * 2);
  bf16*  gB   = (bf16*)take((size_t)MROWS * 2048 * 2);
  bf16*  wInB = (bf16*)take((size_t)4096 * 512 * 2);
  bf16*  wXpB = (bf16*)take((size_t)160 * 2048 * 2);
  bf16*  wDtB = (bf16*)take((size_t)2048 * 32 * 2);
  bf16*  wOpB = (bf16*)take((size_t)512 * 2048 * 2);
  bf16*  wW1B = (bf16*)take((size_t)2048 * 512 * 2);
  bf16*  wW2B = (bf16*)take((size_t)512 * 2048 * 2);

  auto castw = [&](const float* s, bf16* dst, int n) {
    k_cast<<<(n + 255) / 256, 256, 0, stream>>>(s, dst, n);
  };
  castw(in_proj, wInB, 4096 * 512);
  castw(x_proj, wXpB, 160 * 2048);
  castw(dt_w, wDtB, 2048 * 32);
  castw(out_proj, wOpB, 512 * 2048);
  castw(w1, wW1B, 2048 * 512);
  castw(w2, wW2B, 512 * 2048);

  auto gemm = [&](const bf16* A, const bf16* W, int N, int K,
                  const float* bias, const float* resid, int act, float* outF,
                  bf16* outB) {
    dim3 grid(((N / 16) + 7) / 8, MROWS / 32);
    k_gemm_bf16<<<grid, 256, 0, stream>>>(A, W, N, K, bias, resid, act, outF,
                                          outB);
  };

  // 1) LN1 -> bf16
  k_layernorm_cast<<<256, 256, 0, stream>>>(x, ln1_g, ln1_b, yb, 1e-5f);
  // 2) in_proj: xz = y @ Win^T            (2048 x 4096, K=512)
  gemm(yb, wInB, 4096, 512, nullptr, nullptr, 0, xz, nullptr);
  // 3) depthwise conv + SiLU on u half
  k_conv_silu<<<(MROWS * DINQ) / 256, 256, 0, stream>>>(xz, conv_w, conv_b, uF,
                                                        uB);
  // 4) x_proj: x_dbl = u @ Wx^T           (2048 x 160, K=2048)
  gemm(uB, wXpB, 160, 2048, nullptr, nullptr, 0, xdbl, nullptr);
  // 5) dt = softplus(x_dbl[:, :32] @ Wdt^T + dt_b)  (2048 x 2048, K=32)
  k_dtslice<<<(MROWS * DTR) / 256, 256, 0, stream>>>(xdbl, dtin);
  gemm(dtin, wDtB, 2048, 32, dt_b, nullptr, 1, dtv, nullptr);
  // 6) selective scan + gating -> ys bf16
  k_scan<<<512, 256, 0, stream>>>(dtv, xdbl, uF, xz, A_log, Dp, ysB);
  // 7) out_proj + residual x -> x2        (2048 x 512, K=2048)
  gemm(ysB, wOpB, 512, 2048, nullptr, x, 0, x2, nullptr);
  // 8) LN2 -> bf16
  k_layernorm_cast<<<256, 256, 0, stream>>>(x2, ln2_g, ln2_b, hB, 1e-6f);
  // 9) MLP up + exact GELU -> bf16        (2048 x 2048, K=512)
  gemm(hB, wW1B, 2048, 512, b1, nullptr, 2, nullptr, gB);
  // 10) MLP down + bias + residual -> out (2048 x 512, K=2048)
  gemm(gB, wW2B, 512, 2048, b2, x2, 0, (float*)d_out, nullptr);
}
